// LearnAdjacency_66245575573970
// MI455X (gfx1250) — compile-verified, matrix operand order checked by reading, not run on hardware
//
#include <hip/hip_runtime.h>
#include <hip/hip_bf16.h>

typedef __attribute__((ext_vector_type(16))) _Float16 v16h;
typedef __attribute__((ext_vector_type(8)))  float    v8f;

#define LN_EPS 1e-5f
#define SM_EPS 1e-8f

// One block = one output row (b, i): 256 threads = 8 wave32.
// Each wave handles two 16-wide j-groups; per group, 4 accumulating
// v_wmma_f32_16x16x32_f16 contract F=128 of |Xi - Xj| * w (w folded into A,
// B = all-ones so every D column carries the 16 row results).
__global__ __launch_bounds__(256)
void learn_adjacency_row_kernel(const float* __restrict__ X,
                                const float* __restrict__ w,
                                const float* __restrict__ bvec,
                                const float* __restrict__ ln_gamma,
                                const float* __restrict__ ln_beta,
                                float* __restrict__ out) {
    __shared__ float sXi[128];   // X[b, i, :]
    __shared__ float sW[128];    // w[:]
    __shared__ float sY[256];    // post-LN logits for this row
    __shared__ float sRed[256];  // softmax reduction scratch

    const int bid  = blockIdx.x;        // [0, B*N)
    const int b    = bid >> 8;
    const int i    = bid & 255;
    const int tid  = threadIdx.x;
    const int lane = tid & 31;
    const int wave = tid >> 5;

    const float* Xb = X + ((size_t)b << 15);   // b * 256 * 128

    if (tid < 128) sXi[tid] = Xb[i * 128 + tid];
    else           sW[tid - 128] = w[tid - 128];
    __syncthreads();

    const float bias  = bvec[0];
    const float gamma = ln_gamma[0];
    const float beta  = ln_beta[0];

    // B matrix: all ones (layout-invariant; every column of D = row sums).
    v16h ones;
#pragma unroll
    for (int e = 0; e < 16; ++e) ones[e] = (_Float16)1.0f;

    const int m    = lane & 15;   // A-matrix row held by this lane
    const int half = lane >> 4;   // K-half selector (ISA A-layout, 16-bit 16x32)

#pragma unroll
    for (int g = 0; g < 2; ++g) {
        const int j0 = (wave * 2 + g) * 16;     // 16 consecutive j per group
        const int j  = j0 + m;
        const float* Xj = Xb + j * 128;

        v8f acc = {0.f, 0.f, 0.f, 0.f, 0.f, 0.f, 0.f, 0.f};

#pragma unroll
        for (int c = 0; c < 4; ++c) {           // F chunks of 32
            const int f0 = c * 32;
            // element e -> K(e) = (e<8 ? e : e+8) + half*8  (two 8-float runs)
            const int fa = f0 + half * 8;       // K = 0..7   (+half*8)
            const int fb = f0 + 16 + half * 8;  // K = 16..23 (+half*8)
            v16h a;
#pragma unroll
            for (int t = 0; t < 8; ++t) {
                float d1 = fabsf(Xj[fa + t] - sXi[fa + t]) * sW[fa + t];
                float d2 = fabsf(Xj[fb + t] - sXi[fb + t]) * sW[fb + t];
                a[t]     = (_Float16)d1;
                a[8 + t] = (_Float16)d2;
            }
            // D = A * ones + C  (accumulate over F chunks)
            acc = __builtin_amdgcn_wmma_f32_16x16x32_f16(
                false, a, false, ones, (short)0, acc, false, false);
        }

        // Column 0 lanes (lane 0 -> rows 0..7, lane 16 -> rows 8..15) scatter
        // bias+ReLU+LayerNorm(size-1 axis => collapses to beta, kept faithful).
        if ((lane & 15) == 0) {
#pragma unroll
            for (int r = 0; r < 8; ++r) {
                float t  = acc[r] + bias;
                t        = t > 0.0f ? t : 0.0f;           // relu
                float mu = t;                              // mean over 1 elem
                float va = (t - mu) * (t - mu);            // var = 0
                float y  = (t - mu) * rsqrtf(va + LN_EPS) * gamma + beta;
                sY[j0 + half * 8 + r] = y;
            }
        }
    }
    __syncthreads();

    // Row softmax with eps-stabilized denominator over the 256 y values.
    const float v = sY[tid];
    sRed[tid] = v;
    __syncthreads();
#pragma unroll
    for (int s = 128; s > 0; s >>= 1) {
        if (tid < s) sRed[tid] = fmaxf(sRed[tid], sRed[tid + s]);
        __syncthreads();
    }
    const float mx = sRed[0];
    __syncthreads();

    const float e = __expf(v - mx);
    sRed[tid] = e;
    __syncthreads();
#pragma unroll
    for (int s = 128; s > 0; s >>= 1) {
        if (tid < s) sRed[tid] += sRed[tid + s];
        __syncthreads();
    }
    const float denom = sRed[0] + SM_EPS;

    out[((size_t)bid << 8) + tid] = e / denom;
}

extern "C" void kernel_launch(void* const* d_in, const int* in_sizes, int n_in,
                              void* d_out, int out_size, void* d_ws, size_t ws_size,
                              hipStream_t stream) {
    (void)in_sizes; (void)n_in; (void)out_size; (void)d_ws; (void)ws_size;
    const float* X        = (const float*)d_in[0];  // (8, 256, 128)
    const float* w        = (const float*)d_in[1];  // (128,)
    const float* bvec     = (const float*)d_in[2];  // (1,)
    const float* ln_gamma = (const float*)d_in[3];  // (1,)
    const float* ln_beta  = (const float*)d_in[4];  // (1,)
    float* out            = (float*)d_out;          // (8, 256, 256)

    const int rows = 8 * 256;                       // B * N
    learn_adjacency_row_kernel<<<rows, 256, 0, stream>>>(
        X, w, bvec, ln_gamma, ln_beta, out);
}
